// Net_17669495456404
// MI455X (gfx1250) — matrix operand order repaired and checked by reading, not compiled, not run on hardware
//
#include <hip/hip_runtime.h>
#include <hip/hip_bf16.h>
#include <math.h>

// ---------------- types ----------------
typedef __attribute__((ext_vector_type(16))) __bf16        v16bf;
typedef __attribute__((ext_vector_type(8)))  float         v8f;
typedef __attribute__((ext_vector_type(4)))  unsigned int  u32x4;
typedef __attribute__((ext_vector_type(8)))  unsigned int  u32x8;
typedef __attribute__((ext_vector_type(4)))  int           i32x4;

typedef i32x4 __attribute__((address_space(1))) as1_i32x4;
typedef i32x4 __attribute__((address_space(3))) as3_i32x4;

#if defined(__has_builtin)
#  if __has_builtin(__builtin_amdgcn_global_load_async_to_lds_b128) && \
      __has_builtin(__builtin_amdgcn_s_wait_asynccnt)
#    define USE_ASYNC_LDS 1
#  endif
#endif
#ifndef USE_ASYNC_LDS
#  define USE_ASYNC_LDS 0
#endif

#define EPS 1e-5f

__device__ __forceinline__ __bf16 f2bf(float f) {
    unsigned u = __float_as_uint(f);
    unsigned r = (u + 0x7FFFu + ((u >> 16) & 1u)) >> 16;
    unsigned short s = (unsigned short)r;
    return __builtin_bit_cast(__bf16, s);
}

__device__ __forceinline__ void atomicMaxF(float* addr, float val) {
    if (val >= 0.0f) {
        atomicMax((int*)addr, __float_as_int(val));
    } else {
        atomicMin((unsigned int*)addr, (unsigned int)__float_as_int(val));
    }
}

// ---------------- generic fill ----------------
__global__ void fill_f32(float* __restrict__ p, float v, size_t n) {
    size_t i = blockIdx.x * (size_t)blockDim.x + threadIdx.x;
    if (i < n) p[i] = v;
}

// ---------------- weight concat (conv1): [37 x 192] f32 ----------------
__global__ void concat_w1(const float* __restrict__ Wm, const float* __restrict__ bm,
                          const float* __restrict__ We, const float* __restrict__ be,
                          const float* __restrict__ Ws, float* __restrict__ out) {
    int tid = blockIdx.x * blockDim.x + threadIdx.x;
    if (tid >= 37 * 192) return;
    int r = tid / 192, c = tid % 192;
    float v;
    if      (r < 16) v = Wm[r * 192 + c];
    else if (r < 20) v = We[(r - 16) * 192 + c];
    else if (r < 36) v = Ws[(r - 20) * 192 + c];
    else             v = bm[c] + be[c];
    out[tid] = v;
}

// ---------------- weight concat (conv2): [213 x 208] f32 ----------------
__global__ void concat_w2(const float* __restrict__ Wm, const float* __restrict__ bm,
                          const float* __restrict__ We, const float* __restrict__ be,
                          float* __restrict__ out) {
    int tid = blockIdx.x * blockDim.x + threadIdx.x;
    if (tid >= 213 * 208) return;
    int r = tid / 208, c = tid % 208;
    float v;
    if      (r < 208) v = Wm[r * 208 + c];
    else if (r < 212) v = We[(r - 208) * 208 + c];
    else              v = bm[c] + be[c];
    out[tid] = v;
}

// ---------------- pack f32 [K x Nout] into WMMA bf16 B-fragments ----------------
// layout: frag[((ct*nK + kt)*32 + lane)*16 + j], lane%16 = n within col tile,
// k = kt*32 + (lane/16)*16 + j  (contiguous k order, same as the A fragments in aFrag)
__global__ void pack_b_frag(const float* __restrict__ W, __bf16* __restrict__ out,
                            int K, int Nout, int Kpad) {
    size_t tid = blockIdx.x * (size_t)blockDim.x + threadIdx.x;
    size_t total = (size_t)Nout * Kpad;
    if (tid >= total) return;
    int j = (int)(tid & 15);
    size_t t = tid >> 4;
    int lane = (int)(t & 31); t >>= 5;
    int nK = Kpad >> 5;
    int kt = (int)(t % nK);
    int ct = (int)(t / nK);
    int n = ct * 16 + (lane & 15);
    int k = kt * 32 + (lane >> 4) * 16 + j;
    float val = (k < K && n < Nout) ? W[(size_t)k * Nout + n] : 0.0f;
    out[tid] = f2bf(val);
}

// ---------------- edge scatter pass 1: agg_x, agg_e, deg ----------------
__global__ void edge_scatter1(const int* __restrict__ ei, const float* __restrict__ x,
                              const float* __restrict__ ea,
                              float* __restrict__ aggx, float* __restrict__ agge,
                              float* __restrict__ deg, int E) {
    int e = blockIdx.x * blockDim.x + threadIdx.x;
    if (e >= E) return;
    int src = ei[e];
    int dst = ei[E + e];
    const float4* xs = (const float4*)(x + (size_t)src * 16);
    float* ox = aggx + (size_t)dst * 16;
#pragma unroll
    for (int v = 0; v < 4; ++v) {
        float4 q = xs[v];
        atomicAdd(ox + 4 * v + 0, q.x);
        atomicAdd(ox + 4 * v + 1, q.y);
        atomicAdd(ox + 4 * v + 2, q.z);
        atomicAdd(ox + 4 * v + 3, q.w);
    }
    float4 a = *(const float4*)(ea + (size_t)e * 4);
    float* oe = agge + (size_t)dst * 4;
    atomicAdd(oe + 0, a.x);
    atomicAdd(oe + 1, a.y);
    atomicAdd(oe + 2, a.z);
    atomicAdd(oe + 3, a.w);
    atomicAdd(deg + dst, 1.0f);
}

// ---------------- edge scatter pass 2: agg_h1 (192 f32/edge, one wave/edge) ----
__global__ void edge_scatter2(const int* __restrict__ ei, const float* __restrict__ x1,
                              float* __restrict__ aggh, int E) {
    size_t tid = blockIdx.x * (size_t)blockDim.x + threadIdx.x;
    int e = (int)(tid >> 5);
    int lane = (int)(tid & 31);
    if (e >= E) return;
    int src = ei[e];
    int dst = ei[E + e];
    const float* xr = x1 + (size_t)src * 208;   // first 192 cols = h1 part
    float* o = aggh + (size_t)dst * 192;
#pragma unroll
    for (int j = 0; j < 6; ++j) {
        int c = lane + 32 * j;
        atomicAdd(o + c, xr[c]);
    }
}

// ---------------- build Xcat1 bf16 [N x 64] ----------------
__global__ void build_xcat1(const float* __restrict__ aggx, const float* __restrict__ agge,
                            const float* __restrict__ x, const float* __restrict__ deg,
                            __bf16* __restrict__ out, int N) {
    size_t tid = blockIdx.x * (size_t)blockDim.x + threadIdx.x;
    if (tid >= (size_t)N * 64) return;
    int c = (int)(tid & 63);
    int n = (int)(tid >> 6);
    float v;
    if      (c < 16) v = aggx[(size_t)n * 16 + c];
    else if (c < 20) v = agge[(size_t)n * 4 + (c - 16)];
    else if (c < 36) v = x[(size_t)n * 16 + (c - 20)];
    else if (c == 36) v = deg[n];
    else              v = 0.0f;
    out[tid] = f2bf(v);
}

// ---------------- build Xcat2 bf16 [N x 224] ----------------
__global__ void build_xcat2(const float* __restrict__ aggh, const float* __restrict__ aggx,
                            const float* __restrict__ agge, const float* __restrict__ deg,
                            __bf16* __restrict__ out, int N) {
    size_t tid = blockIdx.x * (size_t)blockDim.x + threadIdx.x;
    if (tid >= (size_t)N * 224) return;
    int c = (int)(tid % 224);
    int n = (int)(tid / 224);
    float v;
    if      (c < 192) v = aggh[(size_t)n * 192 + c];
    else if (c < 208) v = aggx[(size_t)n * 16 + (c - 192)];
    else if (c < 212) v = agge[(size_t)n * 4 + (c - 208)];
    else if (c == 212) v = deg[n];
    else               v = 0.0f;
    out[tid] = f2bf(v);
}

// ---------------- WMMA bf16 GEMM ----------------
// C[MxNout] = A[MxKpad] * Bfrag (+bias)(+residual)(+PReLU)
// CONTRACT: M % 16 == 0 (all call sites: 50000, 64). EXEC stays all-ones.
// A tile (16 x Kpad, contiguous in global) is staged to LDS once per block via
// gfx1250 async global->LDS DMA; B fragments stream from global, ping-pong
// buffered (two named buffers -> no register-rotation movs).
__global__ void gemm_bf16_wmma(const __bf16* __restrict__ A, const __bf16* __restrict__ Bf,
                               float* __restrict__ C, const float* __restrict__ bias,
                               const float* __restrict__ residual,
                               const float* __restrict__ alphaPtr,
                               int M, int Nout, int Kpad) {
    extern __shared__ __bf16 sA[];               // [16][Kpad]
    const int lane = threadIdx.x & 31;
    const int wave = threadIdx.x >> 5;           // 8 waves per block
    const int row0 = blockIdx.x * 16;
    const int nCT = Nout >> 4;
    const int nK = Kpad >> 5;
    const int h = lane >> 4;
    const float alpha = alphaPtr ? alphaPtr[0] : 0.0f;

    // ---- stage A tile into LDS (contiguous 16*Kpad bf16 = 2*Kpad 16B chunks) ----
    {
        const __bf16* src = A + (size_t)row0 * Kpad;
        const int chunks = 2 * Kpad;
        for (int i = threadIdx.x; i < chunks; i += 256) {
#if USE_ASYNC_LDS
            __builtin_amdgcn_global_load_async_to_lds_b128(
                (as1_i32x4*)(src + i * 8),
                (as3_i32x4*)(sA + i * 8), 0, 0);
#else
            *(u32x4*)(sA + i * 8) = *(const u32x4*)(src + i * 8);
#endif
        }
#if USE_ASYNC_LDS
        __builtin_amdgcn_s_wait_asynccnt(0);
#endif
        __syncthreads();
    }

    const __bf16* arow = sA + (lane & 15) * Kpad;
    // A fragment from LDS (ds_load_b128 x2) for k-tile kt
    auto aFrag = [&](int kt) -> v16bf {
        const __bf16* ap = arow + kt * 32;
        u32x8 aa;
        aa.lo = *(const u32x4*)(ap + 16 * h);
        aa.hi = *(const u32x4*)(ap + 16 * h + 8);
        return __builtin_bit_cast(v16bf, aa);
    };

    for (int ct = wave; ct < nCT; ct += 8) {
        v8f acc = {0.f, 0.f, 0.f, 0.f, 0.f, 0.f, 0.f, 0.f};
        const __bf16* bbase = Bf + (size_t)(ct * nK) * 512 + lane * 16;  // 32 lanes*16 bf16/kt

        u32x8 b0, b1;
        b0.lo = *(const u32x4*)(bbase);
        b0.hi = *(const u32x4*)(bbase + 8);

        int kt = 0;
        for (; kt + 1 < nK; kt += 2) {
            // load odd buffer for kt+1
            const __bf16* bp1 = bbase + (size_t)(kt + 1) * 512;
            b1.lo = *(const u32x4*)(bp1);
            b1.hi = *(const u32x4*)(bp1 + 8);
            if (kt + 2 < nK)
                __builtin_prefetch(bbase + (size_t)(kt + 2) * 512, 0, 3);  // near-scope

            acc = __builtin_amdgcn_wmma_f32_16x16x32_bf16(
                false, aFrag(kt), false, __builtin_bit_cast(v16bf, b0),
                (short)0, acc, false, false);

            // refill even buffer for kt+2 while wmma(kt+1) runs
            if (kt + 2 < nK) {
                const __bf16* bp2 = bbase + (size_t)(kt + 2) * 512;
                b0.lo = *(const u32x4*)(bp2);
                b0.hi = *(const u32x4*)(bp2 + 8);
            }
            acc = __builtin_amdgcn_wmma_f32_16x16x32_bf16(
                false, aFrag(kt + 1), false, __builtin_bit_cast(v16bf, b1),
                (short)0, acc, false, false);
        }
        if (kt < nK) {   // odd-nK tail (b0 was refilled in the last pair)
            acc = __builtin_amdgcn_wmma_f32_16x16x32_bf16(
                false, aFrag(kt), false, __builtin_bit_cast(v16bf, b0),
                (short)0, acc, false, false);
        }

        // epilogue: vgpr r -> (m = h*8 + r, n = ct*16 + lane%16)
#pragma unroll
        for (int r = 0; r < 8; ++r) {
            int mm = row0 + h * 8 + r;
            int nn = ct * 16 + (lane & 15);
            float v = acc[r];
            if (bias)     v += bias[nn];
            if (residual) v += residual[(size_t)mm * Nout + nn];
            if (alphaPtr) v = (v >= 0.0f) ? v : alpha * v;
            C[(size_t)mm * Nout + nn] = v;
        }
    }
}

// ---------------- BN stats: one block (256 thr) per channel -> scale/shift ----
__global__ void bn_stats(const float* __restrict__ Hm, int N, int C,
                         const float* __restrict__ g, const float* __restrict__ be,
                         float* __restrict__ scale, float* __restrict__ shift) {
    int c = blockIdx.x;
    float s = 0.0f, sq = 0.0f;
    for (int r = threadIdx.x; r < N; r += 256) {
        float v = Hm[(size_t)r * C + c];
        s += v; sq += v * v;
    }
    __shared__ float ss[256], sqq[256];
    ss[threadIdx.x] = s; sqq[threadIdx.x] = sq;
    __syncthreads();
    for (int o = 128; o > 0; o >>= 1) {
        if ((int)threadIdx.x < o) {
            ss[threadIdx.x]  += ss[threadIdx.x + o];
            sqq[threadIdx.x] += sqq[threadIdx.x + o];
        }
        __syncthreads();
    }
    if (threadIdx.x == 0) {
        float mean = ss[0] / (float)N;
        float var  = sqq[0] / (float)N - mean * mean;
        float sc = g[c] * rsqrtf(var + EPS);
        scale[c] = sc;
        shift[c] = be[c] - mean * sc;
    }
}

// ---------------- build x1 f32 [N x 208] = [BN(h1), x] ----------------
__global__ void build_x1(const float* __restrict__ h1, const float* __restrict__ scale,
                         const float* __restrict__ shift, const float* __restrict__ x,
                         float* __restrict__ x1, int N) {
    size_t tid = blockIdx.x * (size_t)blockDim.x + threadIdx.x;
    if (tid >= (size_t)N * 208) return;
    int c = (int)(tid % 208);
    int n = (int)(tid / 208);
    float v = (c < 192) ? (h1[(size_t)n * 192 + c] * scale[c] + shift[c])
                        : x[(size_t)n * 16 + (c - 192)];
    x1[tid] = v;
}

// ---------------- node counts per graph ----------------
__global__ void count_nodes(const int* __restrict__ batch, float* __restrict__ cnt, int N) {
    int n = blockIdx.x * blockDim.x + threadIdx.x;
    if (n < N) atomicAdd(cnt + batch[n], 1.0f);
}

// ---------------- pooling: sum + max over x2 = [BN(h2), x1] ----------------
__global__ void pool_kernel(const float* __restrict__ h2, const float* __restrict__ x1,
                            const float* __restrict__ scale, const float* __restrict__ shift,
                            const int* __restrict__ batch,
                            float* __restrict__ psum, float* __restrict__ pmax, int N) {
    size_t tid = blockIdx.x * (size_t)blockDim.x + threadIdx.x;
    if (tid >= (size_t)N * 416) return;
    int c = (int)(tid % 416);
    int n = (int)(tid / 416);
    float v = (c < 208) ? (h2[(size_t)n * 208 + c] * scale[c] + shift[c])
                        : x1[(size_t)n * 208 + (c - 208)];
    int b = batch[n];
    atomicAdd(psum + (size_t)b * 416 + c, v);
    atomicMaxF(pmax + (size_t)b * 416 + c, v);
}

// ---------------- pooled -> bf16 A [64 x 1248] : [sum, max, mean] ----------------
__global__ void pooled_build(const float* __restrict__ psum, const float* __restrict__ pmax,
                             const float* __restrict__ cnt, __bf16* __restrict__ A, int B) {
    int tid = blockIdx.x * blockDim.x + threadIdx.x;
    if (tid >= B * 1248) return;
    int c = tid % 1248;
    int b = tid / 1248;
    float v;
    if      (c < 416) v = psum[b * 416 + c];
    else if (c < 832) v = pmax[b * 416 + (c - 416)];
    else              v = psum[b * 416 + (c - 832)] / fmaxf(cnt[b], 1.0f);
    A[tid] = f2bf(v);
}

// ---------------- final head: hid[64x624] @ W2[624x2] + b2, log_softmax --------
__global__ void head_kernel(const float* __restrict__ hid, const float* __restrict__ W2,
                            const float* __restrict__ b2, float* __restrict__ out, int B) {
    int b = blockIdx.x * blockDim.x + threadIdx.x;
    if (b >= B) return;
    float l0 = b2[0], l1 = b2[1];
    const float* h = hid + (size_t)b * 624;
    for (int k = 0; k < 624; ++k) {
        float v = h[k];
        l0 += v * W2[k * 2 + 0];
        l1 += v * W2[k * 2 + 1];
    }
    float mx = fmaxf(l0, l1);
    float lse = mx + logf(expf(l0 - mx) + expf(l1 - mx));
    out[b * 2 + 0] = l0 - lse;
    out[b * 2 + 1] = l1 - lse;
}

// ============================================================================
extern "C" void kernel_launch(void* const* d_in, const int* in_sizes, int n_in,
                              void* d_out, int out_size, void* d_ws, size_t ws_size,
                              hipStream_t stream) {
    const float* x       = (const float*)d_in[0];
    const int*   ei      = (const int*)d_in[1];
    const float* ea      = (const float*)d_in[2];
    const int*   batch   = (const int*)d_in[3];
    const float* W_msg1  = (const float*)d_in[4];
    const float* b_msg1  = (const float*)d_in[5];
    const float* W_edge1 = (const float*)d_in[6];
    const float* b_edge1 = (const float*)d_in[7];
    const float* W_self1 = (const float*)d_in[8];
    const float* b_self1 = (const float*)d_in[9];
    const float* g1      = (const float*)d_in[10];
    const float* be1     = (const float*)d_in[11];
    const float* W_msg2  = (const float*)d_in[12];
    const float* b_msg2  = (const float*)d_in[13];
    const float* W_edge2 = (const float*)d_in[14];
    const float* b_edge2 = (const float*)d_in[15];
    const float* g2      = (const float*)d_in[16];
    const float* be2     = (const float*)d_in[17];
    const float* W_l1    = (const float*)d_in[18];
    const float* b_l1    = (const float*)d_in[19];
    const float* alpha   = (const float*)d_in[20];
    const float* W_l2    = (const float*)d_in[21];
    const float* b_l2    = (const float*)d_in[22];

    const int N = in_sizes[0] / 16;
    const int E = in_sizes[2] / 4;
    const int B = 64;

    // ---- workspace carve-out (256B aligned) ----
    char* ws = (char*)d_ws;
    size_t off = 0;
    auto alloc = [&](size_t bytes) -> void* {
        off = (off + 255) & ~(size_t)255;
        void* p = ws + off;
        off += bytes;
        return p;
    };
    float*  aggx    = (float*)alloc((size_t)N * 16 * 4);
    float*  agge    = (float*)alloc((size_t)N * 4 * 4);
    float*  deg     = (float*)alloc((size_t)N * 4);
    float*  wcat1   = (float*)alloc(37 * 192 * 4);
    __bf16* wfrag1  = (__bf16*)alloc((size_t)64 * 192 * 2);
    __bf16* xcat1   = (__bf16*)alloc((size_t)N * 64 * 2);
    float*  h1pre   = (float*)alloc((size_t)N * 192 * 4);
    float*  scale1  = (float*)alloc(192 * 4);
    float*  shift1  = (float*)alloc(192 * 4);
    float*  x1      = (float*)alloc((size_t)N * 208 * 4);
    float*  aggh    = (float*)alloc((size_t)N * 192 * 4);
    float*  wcat2   = (float*)alloc(213 * 208 * 4);
    __bf16* wfrag2  = (__bf16*)alloc((size_t)224 * 208 * 2);
    __bf16* xcat2   = (__bf16*)alloc((size_t)N * 224 * 2);
    float*  h2pre   = (float*)alloc((size_t)N * 208 * 4);
    float*  scale2  = (float*)alloc(208 * 4);
    float*  shift2  = (float*)alloc(208 * 4);
    float*  psum    = (float*)alloc((size_t)B * 416 * 4);
    float*  pmax    = (float*)alloc((size_t)B * 416 * 4);
    float*  cnt     = (float*)alloc((size_t)B * 4);
    __bf16* pooledA = (__bf16*)alloc((size_t)B * 1248 * 2);
    __bf16* wfragL1 = (__bf16*)alloc((size_t)1248 * 624 * 2);
    float*  hid     = (float*)alloc((size_t)B * 624 * 4);

    const int T = 256;
    auto cdiv = [](size_t a, size_t b) { return (unsigned)((a + b - 1) / b); };

    // ---- zero / init accumulators (in-stream, graph-capturable) ----
    (void)hipMemsetAsync(aggx, 0, (size_t)N * 16 * 4, stream);
    (void)hipMemsetAsync(agge, 0, (size_t)N * 4 * 4, stream);
    (void)hipMemsetAsync(deg,  0, (size_t)N * 4, stream);
    (void)hipMemsetAsync(aggh, 0, (size_t)N * 192 * 4, stream);
    (void)hipMemsetAsync(psum, 0, (size_t)B * 416 * 4, stream);
    (void)hipMemsetAsync(cnt,  0, (size_t)B * 4, stream);
    fill_f32<<<cdiv((size_t)B * 416, T), T, 0, stream>>>(pmax, -INFINITY, (size_t)B * 416);

    // ---- pack weights into WMMA B-fragments ----
    concat_w1<<<cdiv(37 * 192, T), T, 0, stream>>>(W_msg1, b_msg1, W_edge1, b_edge1, W_self1, wcat1);
    pack_b_frag<<<cdiv((size_t)64 * 192, T), T, 0, stream>>>(wcat1, wfrag1, 37, 192, 64);
    concat_w2<<<cdiv(213 * 208, T), T, 0, stream>>>(W_msg2, b_msg2, W_edge2, b_edge2, wcat2);
    pack_b_frag<<<cdiv((size_t)224 * 208, T), T, 0, stream>>>(wcat2, wfrag2, 213, 208, 224);
    pack_b_frag<<<cdiv((size_t)1248 * 624, T), T, 0, stream>>>(W_l1, wfragL1, 1248, 624, 1248);

    // ---- conv1: edge aggregation -> fused node GEMM -> BN params -> x1 ----
    edge_scatter1<<<cdiv(E, T), T, 0, stream>>>(ei, x, ea, aggx, agge, deg, E);
    build_xcat1<<<cdiv((size_t)N * 64, T), T, 0, stream>>>(aggx, agge, x, deg, xcat1, N);
    gemm_bf16_wmma<<<cdiv(N, 16), 256, 16 * 64 * 2, stream>>>(
        xcat1, wfrag1, h1pre, b_self1, nullptr, nullptr, N, 192, 64);
    bn_stats<<<192, 256, 0, stream>>>(h1pre, N, 192, g1, be1, scale1, shift1);
    build_x1<<<cdiv((size_t)N * 208, T), T, 0, stream>>>(h1pre, scale1, shift1, x, x1, N);

    // ---- conv2: aggregate BN'd h1 -> fused node GEMM (+x1 residual) -> BN params
    edge_scatter2<<<cdiv((size_t)E * 32, T), T, 0, stream>>>(ei, x1, aggh, E);
    build_xcat2<<<cdiv((size_t)N * 224, T), T, 0, stream>>>(aggh, aggx, agge, deg, xcat2, N);
    gemm_bf16_wmma<<<cdiv(N, 16), 256, 16 * 224 * 2, stream>>>(
        xcat2, wfrag2, h2pre, nullptr, x1, nullptr, N, 208, 224);
    bn_stats<<<208, 256, 0, stream>>>(h2pre, N, 208, g2, be2, scale2, shift2);

    // ---- pooling (sum/max/mean) -> MLP head ----
    count_nodes<<<cdiv(N, T), T, 0, stream>>>(batch, cnt, N);
    pool_kernel<<<cdiv((size_t)N * 416, T), T, 0, stream>>>(h2pre, x1, scale2, shift2,
                                                            batch, psum, pmax, N);
    pooled_build<<<cdiv((size_t)B * 1248, T), T, 0, stream>>>(psum, pmax, cnt, pooledA, B);
    gemm_bf16_wmma<<<cdiv(B, 16), 256, 16 * 1248 * 2, stream>>>(
        pooledA, wfragL1, hid, b_l1, nullptr, alpha, B, 624, 1248);
    head_kernel<<<1, 64, 0, stream>>>(hid, W_l2, b_l2, (float*)d_out, B);
}
